// seq2seq_36112085025108
// MI455X (gfx1250) — matrix operand order repaired and checked by reading, not compiled, not run on hardware
//
#include <hip/hip_runtime.h>
#include <cstdint>
#include <cstddef>

// Problem dims (match reference)
#define TSs 50
#define TTs 48
#define Bs  64
#define Es  1024
#define Hs  1024
#define Gs  4096      // 4*H gates
#define Vs  32000

typedef __attribute__((ext_vector_type(16))) __bf16 v16bf;
typedef __attribute__((ext_vector_type(8)))  __bf16 v8bf;
typedef __attribute__((ext_vector_type(8)))  float  v8f;

typedef unsigned short u16;
typedef long long ll;

__device__ __forceinline__ u16 f2bf(float f) {
  unsigned int u = __float_as_uint(f);
  u += 0x7FFFu + ((u >> 16) & 1u);          // round-to-nearest-even
  return (u16)(u >> 16);
}
__device__ __forceinline__ float sigm(float x) { return 1.0f / (1.0f + __expf(-x)); }

// Assemble the 16-bit A fragment for v_wmma_f32_16x16x32_bf16 from a row-major
// row pointer already offset for this lane: elements {0..7} = K+0..7(+8 for
// upper half-wave), {8..15} = K+16..23(+8)  (ISA 7.12.2 16-bit A layout).
__device__ __forceinline__ v16bf afrag(const __bf16* p) {
  v8bf lo = *(const v8bf*)p;
  v8bf hi = *(const v8bf*)(p + 16);
  v16bf r;
  #pragma unroll
  for (int i = 0; i < 8; ++i) { r[i] = lo[i]; r[i + 8] = hi[i]; }
  return r;
}

// ---------------------------------------------------------------------------
// Elementwise helpers
// ---------------------------------------------------------------------------
__global__ void k_f32_to_bf16(const float* __restrict__ in, u16* __restrict__ out, ll n) {
  ll i = (ll)blockIdx.x * blockDim.x + threadIdx.x;
  if (i < n) out[i] = f2bf(in[i]);
}
__global__ void k_add_bias2(const float* __restrict__ a, const float* __restrict__ b,
                            float* __restrict__ out, int n) {
  int i = blockIdx.x * blockDim.x + threadIdx.x;
  if (i < n) out[i] = a[i] + b[i];
}
__global__ void k_gather_bf16(const float* __restrict__ table, const int* __restrict__ idx,
                              u16* __restrict__ out, ll n) { // dim = 1024
  ll i = (ll)blockIdx.x * blockDim.x + threadIdx.x;
  if (i < n) {
    ll tok = i >> 10; int d = (int)(i & 1023);
    out[i] = f2bf(table[(size_t)idx[tok] * 1024 + d]);
  }
}
__global__ void k_zero_f32(float* p, int n) {
  int i = blockIdx.x * blockDim.x + threadIdx.x; if (i < n) p[i] = 0.0f;
}
__global__ void k_zero_u16(u16* p, int n) {
  int i = blockIdx.x * blockDim.x + threadIdx.x; if (i < n) p[i] = 0;
}
__global__ void k_relu_bf16(const float* __restrict__ in, u16* __restrict__ out, int n) {
  int i = blockIdx.x * blockDim.x + threadIdx.x;
  if (i < n) out[i] = f2bf(fmaxf(in[i], 0.0f));
}

// ---------------------------------------------------------------------------
// WMMA bf16 GEMM:  C[M x N] = A0[M x K0] * W0[N x K0]^T (+ A1*W1^T) (+Cadd)(+bias)
// Each wave computes a 64x16 strip: 4 M-subtiles share ONE B (weight) fragment
// per K-block -> weights (the dominant stream) are fetched exactly once per
// N-tile, and the 4 independent accumulator chains give the scheduler 4
// independent v_wmma per K-block. 4 waves per block = 4 adjacent N tiles.
// Requires: M % 64 == 0, N % 64 == 0, K0/K1 % 32 == 0.
// ---------------------------------------------------------------------------
__global__ void __launch_bounds__(128)
k_gemm_wmma_bf16(const u16* __restrict__ A0, int lda0,
                 const u16* __restrict__ W0, int ldw0, int K0,
                 const u16* __restrict__ A1, int lda1,
                 const u16* __restrict__ W1, int ldw1, int K1,
                 const float* __restrict__ Cadd, int ldcadd,
                 const float* __restrict__ bias,
                 float* __restrict__ Cf, ll ldcf, ll cstride,
                 u16* __restrict__ Cbf, int ldcbf)
{
  const int lane = threadIdx.x & 31;
  const int half = lane >> 4;            // 0: lanes 0-15, 1: lanes 16-31
  const int l    = lane & 15;
  const int nbase = (blockIdx.x * blockDim.y + threadIdx.y) * 16;
  const int mgrp  = blockIdx.y * 64;

  v8f acc0 = {}, acc1 = {}, acc2 = {}, acc3 = {};

  {
    const size_t rstep = (size_t)16 * lda0;             // next M subtile
    const __bf16* pa = (const __bf16*)(A0 + (size_t)(mgrp + l) * lda0 + half * 8);
    const __bf16* pb = (const __bf16*)(W0 + (size_t)(nbase + l) * ldw0 + half * 8);
    for (int k = 0; k < K0; k += 32, pa += 32, pb += 32) {
      v16bf b  = afrag(pb);                             // same K layout as A: W row = B column
      if (k + 64 < K0) __builtin_prefetch((const void*)(pb + 64), 0, 2);
      v16bf a0 = afrag(pa);
      v16bf a1 = afrag(pa + rstep);
      v16bf a2 = afrag(pa + 2 * rstep);
      v16bf a3 = afrag(pa + 3 * rstep);
      acc0 = __builtin_amdgcn_wmma_f32_16x16x32_bf16(false, a0, false, b, (short)0, acc0, false, false);
      acc1 = __builtin_amdgcn_wmma_f32_16x16x32_bf16(false, a1, false, b, (short)0, acc1, false, false);
      acc2 = __builtin_amdgcn_wmma_f32_16x16x32_bf16(false, a2, false, b, (short)0, acc2, false, false);
      acc3 = __builtin_amdgcn_wmma_f32_16x16x32_bf16(false, a3, false, b, (short)0, acc3, false, false);
    }
  }
  if (A1) {  // uniform branch: EXEC stays all-ones around WMMA
    const size_t rstep = (size_t)16 * lda1;
    const __bf16* pa = (const __bf16*)(A1 + (size_t)(mgrp + l) * lda1 + half * 8);
    const __bf16* pb = (const __bf16*)(W1 + (size_t)(nbase + l) * ldw1 + half * 8);
    for (int k = 0; k < K1; k += 32, pa += 32, pb += 32) {
      v16bf b  = afrag(pb);
      if (k + 64 < K1) __builtin_prefetch((const void*)(pb + 64), 0, 2);
      v16bf a0 = afrag(pa);
      v16bf a1 = afrag(pa + rstep);
      v16bf a2 = afrag(pa + 2 * rstep);
      v16bf a3 = afrag(pa + 3 * rstep);
      acc0 = __builtin_amdgcn_wmma_f32_16x16x32_bf16(false, a0, false, b, (short)0, acc0, false, false);
      acc1 = __builtin_amdgcn_wmma_f32_16x16x32_bf16(false, a1, false, b, (short)0, acc1, false, false);
      acc2 = __builtin_amdgcn_wmma_f32_16x16x32_bf16(false, a2, false, b, (short)0, acc2, false, false);
      acc3 = __builtin_amdgcn_wmma_f32_16x16x32_bf16(false, a3, false, b, (short)0, acc3, false, false);
    }
  }

  // C/D layout: element r of v8f -> (M = subtile + half*8 + r, N = nbase + l)
  const int n = nbase + l;
  const float bb = bias ? bias[n] : 0.0f;
  #pragma unroll
  for (int g = 0; g < 4; ++g) {
    v8f acc = (g == 0) ? acc0 : (g == 1) ? acc1 : (g == 2) ? acc2 : acc3;
    #pragma unroll
    for (int r = 0; r < 8; ++r) {
      const int m = mgrp + g * 16 + half * 8 + r;
      float v = acc[r] + bb;
      if (Cadd) v += Cadd[(size_t)m * ldcadd + n];
      if (Cf)   Cf[(size_t)m * ldcf + (size_t)n * cstride] = v;
      if (Cbf)  Cbf[(size_t)m * ldcbf + n] = f2bf(v);
    }
  }
}

// ---------------------------------------------------------------------------
// LSTM cell: gates (B x 4H, order i,f,g,o) -> update h,c with optional mask;
// writes bf16 h for the next recurrent GEMM, optional masked outputs.
// ---------------------------------------------------------------------------
__global__ void k_lstm_cell(const float* __restrict__ gates,
                            const int* __restrict__ src_len, int t,
                            float* __restrict__ h, float* __restrict__ c,
                            u16* __restrict__ h_bf,
                            float* __restrict__ out_f32, u16* __restrict__ out_bf,
                            ll out_row_stride)
{
  int i = blockIdx.x * blockDim.x + threadIdx.x;
  if (i >= Bs * Hs) return;
  int b = i >> 10, d = i & 1023;
  const float* g = gates + (size_t)b * Gs;
  float ig = sigm(g[d]);
  float fg = sigm(g[Hs + d]);
  float gg = tanhf(g[2 * Hs + d]);
  float og = sigm(g[3 * Hs + d]);
  float cn = fg * c[i] + ig * gg;
  float hn = og * tanhf(cn);
  bool valid = src_len ? (t < src_len[b]) : true;
  float hw = valid ? hn : h[i];
  float cw = valid ? cn : c[i];
  h[i] = hw; c[i] = cw; h_bf[i] = f2bf(hw);
  if (out_f32) {
    float ov = valid ? hn : 0.0f;
    out_f32[(size_t)b * out_row_stride + d] = ov;
    out_bf [(size_t)b * out_row_stride + d] = f2bf(ov);
  }
}

// scores[b,t] = dot(attn_proj[b,t,:], h[b,:]); one wave per (t,b)
__global__ void k_scores(const float* __restrict__ attnp, const float* __restrict__ h,
                         float* __restrict__ scores)
{
  int t = blockIdx.x, b = blockIdx.y, lane = threadIdx.x;
  const float* ap = attnp + ((size_t)b * TSs + t) * Hs;
  const float* hb = h + (size_t)b * Hs;
  float s = 0.0f;
  for (int d = lane; d < Hs; d += 32) s += ap[d] * hb[d];
  #pragma unroll
  for (int off = 16; off; off >>= 1) s += __shfl_xor(s, off, 32);
  if (lane == 0) scores[b * TSs + t] = s;
}

// masked softmax over T=50 per batch row (block of 64 threads)
__global__ void k_softmax_mask(const float* __restrict__ scores,
                               const int* __restrict__ src_len,
                               float* __restrict__ wts)
{
  __shared__ float sh[64];
  int b = blockIdx.x, t = threadIdx.x;
  int len = src_len[b];
  bool ok = (t < TSs) && (t < len);
  float v = ok ? scores[b * TSs + t] : -3.4e38f;
  sh[t] = v; __syncthreads();
  for (int s = 32; s > 0; s >>= 1) { if (t < s) sh[t] = fmaxf(sh[t], sh[t + s]); __syncthreads(); }
  float mx = sh[0]; __syncthreads();
  float e = ok ? __expf(v - mx) : 0.0f;
  sh[t] = e; __syncthreads();
  for (int s = 32; s > 0; s >>= 1) { if (t < s) sh[t] += sh[t + s]; __syncthreads(); }
  float inv = 1.0f / sh[0];
  if (t < TSs) wts[b * TSs + t] = e * inv;
}

// ctx[b,d] = sum_t w[b,t] * enc[b,t,d]  (d over 2H), bf16 out for next GEMM
__global__ void k_ctx(const float* __restrict__ wts, const float* __restrict__ encf,
                      u16* __restrict__ ctxbf)
{
  int i = blockIdx.x * blockDim.x + threadIdx.x;
  if (i >= Bs * 2 * Hs) return;
  int b = i >> 11, d = i & 2047;
  const float* e = encf + (size_t)b * TSs * 2 * Hs + d;
  const float* w = wts + b * TSs;
  float s = 0.0f;
  #pragma unroll 5
  for (int t = 0; t < TSs; ++t) s += w[t] * e[(size_t)t * 2 * Hs];
  ctxbf[i] = f2bf(s);
}

// ---------------------------------------------------------------------------
// Host side
// ---------------------------------------------------------------------------
static inline void gemm(hipStream_t s, int M, int N,
                        const u16* A0, int lda0, const u16* W0, int ldw0, int K0,
                        const u16* A1, int lda1, const u16* W1, int ldw1, int K1,
                        const float* Cadd, int ldcadd, const float* bias,
                        float* Cf, ll ldcf, ll cstride, u16* Cbf, int ldcbf)
{
  dim3 blk(32, 4), grd(N / 64, M / 64);
  k_gemm_wmma_bf16<<<grd, blk, 0, s>>>(A0, lda0, W0, ldw0, K0, A1, lda1, W1, ldw1, K1,
                                       Cadd, ldcadd, bias, Cf, ldcf, cstride, Cbf, ldcbf);
}
static inline int gblk(ll n) { return (int)((n + 255) / 256); }

extern "C" void kernel_launch(void* const* d_in, const int* in_sizes, int n_in,
                              void* d_out, int out_size, void* d_ws, size_t ws_size,
                              hipStream_t stream)
{
  (void)in_sizes; (void)n_in; (void)out_size; (void)ws_size;
  const int*   src     = (const int*)  d_in[0];
  const int*   src_len = (const int*)  d_in[1];
  const int*   trg     = (const int*)  d_in[2];
  const float* enc_emb = (const float*)d_in[3];
  const float* Wih_f   = (const float*)d_in[4];
  const float* Whh_f   = (const float*)d_in[5];
  const float* bih_f   = (const float*)d_in[6];
  const float* bhh_f   = (const float*)d_in[7];
  const float* Wih_b   = (const float*)d_in[8];
  const float* Whh_b   = (const float*)d_in[9];
  const float* bih_b   = (const float*)d_in[10];
  const float* bhh_b   = (const float*)d_in[11];
  const float* Wfch    = (const float*)d_in[12];
  const float* bfch    = (const float*)d_in[13];
  const float* Wfcc    = (const float*)d_in[14];
  const float* bfcc    = (const float*)d_in[15];
  const float* dec_emb = (const float*)d_in[16];
  const float* Wih_d   = (const float*)d_in[17];
  const float* Whh_d   = (const float*)d_in[18];
  const float* bih_d   = (const float*)d_in[19];
  const float* bhh_d   = (const float*)d_in[20];
  const float* Wattn   = (const float*)d_in[21];
  const float* battn   = (const float*)d_in[22];
  const float* Wcomb   = (const float*)d_in[23];
  const float* bcomb   = (const float*)d_in[24];
  const float* Wout    = (const float*)d_in[25];
  const float* bout    = (const float*)d_in[26];
  float* out = (float*)d_out;

  char* ws = (char*)d_ws;
  size_t off = 0;
#define ALLOC(type, name, count) \
  type* name = (type*)(ws + off); \
  off = (off + (size_t)(count) * sizeof(type) + 255) & ~(size_t)255;

  // bf16 weights (L2-resident working set)
  ALLOC(u16, wWihF, (size_t)Gs * Es)
  ALLOC(u16, wWhhF, (size_t)Gs * Hs)
  ALLOC(u16, wWihB, (size_t)Gs * Es)
  ALLOC(u16, wWhhB, (size_t)Gs * Hs)
  ALLOC(u16, wWihD, (size_t)Gs * 2 * Hs)
  ALLOC(u16, wWhhD, (size_t)Gs * Hs)
  ALLOC(u16, wWattn, (size_t)Hs * 2 * Hs)
  ALLOC(u16, wWcomb, (size_t)Hs * 3 * Hs)
  ALLOC(u16, wWfch, (size_t)Hs * 2 * Hs)
  ALLOC(u16, wWfcc, (size_t)Hs * 2 * Hs)
  ALLOC(u16, wWout, (size_t)Vs * Hs)
  // combined biases
  ALLOC(float, biasF, Gs)
  ALLOC(float, biasB, Gs)
  ALLOC(float, biasD, Gs)
  // embeddings (bf16)
  ALLOC(u16, embE, (size_t)TSs * Bs * Es)
  ALLOC(u16, embD, (size_t)(TTs - 1) * Bs * Hs)
  // precomputed input projections (f32)
  ALLOC(float, XPf, (size_t)TSs * Bs * Gs)
  ALLOC(float, XPb, (size_t)TSs * Bs * Gs)
  ALLOC(float, XPd, (size_t)(TTs - 1) * Bs * Gs)
  // encoder outputs (b, t, 2H)
  ALLOC(float, encf, (size_t)Bs * TSs * 2 * Hs)
  ALLOC(u16,   encbf, (size_t)Bs * TSs * 2 * Hs)
  ALLOC(float, attnp, (size_t)Bs * TSs * Hs)
  // per-step buffers
  ALLOC(float, gates, (size_t)Bs * Gs)
  ALLOC(float, hF, Bs * Hs)  ALLOC(float, cF, Bs * Hs)
  ALLOC(float, hB, Bs * Hs)  ALLOC(float, cB, Bs * Hs)
  ALLOC(float, hD, Bs * Hs)  ALLOC(float, cD, Bs * Hs)
  ALLOC(u16, hFbf, Bs * Hs)  ALLOC(u16, hBbf, Bs * Hs)  ALLOC(u16, hDbf, Bs * Hs)
  ALLOC(u16, doutbf, Bs * Hs)  ALLOC(u16, doutrbf, Bs * Hs)
  ALLOC(u16, ctxbf, Bs * 2 * Hs)
  ALLOC(float, doutf, Bs * Hs)
  ALLOC(float, scores, Bs * TSs)
  ALLOC(float, wts, Bs * TSs)
#undef ALLOC

  // 1) weights -> bf16
  struct { const float* s; u16* d; ll n; } cv[] = {
    {Wih_f, wWihF, (ll)Gs * Es}, {Whh_f, wWhhF, (ll)Gs * Hs},
    {Wih_b, wWihB, (ll)Gs * Es}, {Whh_b, wWhhB, (ll)Gs * Hs},
    {Wih_d, wWihD, (ll)Gs * 2 * Hs}, {Whh_d, wWhhD, (ll)Gs * Hs},
    {Wattn, wWattn, (ll)Hs * 2 * Hs}, {Wcomb, wWcomb, (ll)Hs * 3 * Hs},
    {Wfch, wWfch, (ll)Hs * 2 * Hs}, {Wfcc, wWfcc, (ll)Hs * 2 * Hs},
    {Wout, wWout, (ll)Vs * Hs},
  };
  for (auto& c : cv) k_f32_to_bf16<<<gblk(c.n), 256, 0, stream>>>(c.s, c.d, c.n);

  // 2) combined biases
  k_add_bias2<<<gblk(Gs), 256, 0, stream>>>(bih_f, bhh_f, biasF, Gs);
  k_add_bias2<<<gblk(Gs), 256, 0, stream>>>(bih_b, bhh_b, biasB, Gs);
  k_add_bias2<<<gblk(Gs), 256, 0, stream>>>(bih_d, bhh_d, biasD, Gs);

  // 3) embedding gathers (bf16)
  k_gather_bf16<<<gblk((ll)TSs * Bs * Es), 256, 0, stream>>>(enc_emb, src, embE, (ll)TSs * Bs * Es);
  k_gather_bf16<<<gblk((ll)(TTs - 1) * Bs * Hs), 256, 0, stream>>>(dec_emb, trg, embD, (ll)(TTs - 1) * Bs * Hs);

  // 4) init recurrent state
  k_zero_f32<<<gblk(Bs * Hs), 256, 0, stream>>>(hF, Bs * Hs);
  k_zero_f32<<<gblk(Bs * Hs), 256, 0, stream>>>(cF, Bs * Hs);
  k_zero_f32<<<gblk(Bs * Hs), 256, 0, stream>>>(hB, Bs * Hs);
  k_zero_f32<<<gblk(Bs * Hs), 256, 0, stream>>>(cB, Bs * Hs);
  k_zero_u16<<<gblk(Bs * Hs), 256, 0, stream>>>(hFbf, Bs * Hs);
  k_zero_u16<<<gblk(Bs * Hs), 256, 0, stream>>>(hBbf, Bs * Hs);
  k_zero_u16<<<gblk(Bs * Hs), 256, 0, stream>>>(doutbf, Bs * Hs);

  // 5) hoisted input projections: XP = emb @ Wih^T + (bih+bhh)  (big parallel GEMMs)
  gemm(stream, TSs * Bs, Gs, embE, Es, wWihF, Es, Es, nullptr, 0, nullptr, 0, 0,
       nullptr, 0, biasF, XPf, Gs, 1, nullptr, 0);
  gemm(stream, TSs * Bs, Gs, embE, Es, wWihB, Es, Es, nullptr, 0, nullptr, 0, 0,
       nullptr, 0, biasB, XPb, Gs, 1, nullptr, 0);
  gemm(stream, (TTs - 1) * Bs, Gs, embD, Hs, wWihD, 2 * Hs, Hs, nullptr, 0, nullptr, 0, 0,
       nullptr, 0, biasD, XPd, Gs, 1, nullptr, 0);

  const int cellBlocks = gblk(Bs * Hs);

  // 6) forward encoder LSTM
  for (int t = 0; t < TSs; ++t) {
    gemm(stream, Bs, Gs, hFbf, Hs, wWhhF, Hs, Hs, nullptr, 0, nullptr, 0, 0,
         XPf + (size_t)t * Bs * Gs, Gs, nullptr, gates, Gs, 1, nullptr, 0);
    k_lstm_cell<<<cellBlocks, 256, 0, stream>>>(gates, src_len, t, hF, cF, hFbf,
        encf + (size_t)t * 2 * Hs, encbf + (size_t)t * 2 * Hs, (ll)TSs * 2 * Hs);
  }
  // 7) backward encoder LSTM (t descending, mask m[t], store at t)
  for (int t = TSs - 1; t >= 0; --t) {
    gemm(stream, Bs, Gs, hBbf, Hs, wWhhB, Hs, Hs, nullptr, 0, nullptr, 0, 0,
         XPb + (size_t)t * Bs * Gs, Gs, nullptr, gates, Gs, 1, nullptr, 0);
    k_lstm_cell<<<cellBlocks, 256, 0, stream>>>(gates, src_len, t, hB, cB, hBbf,
        encf + (size_t)t * 2 * Hs + Hs, encbf + (size_t)t * 2 * Hs + Hs, (ll)TSs * 2 * Hs);
  }

  // 8) decoder init: hidden = [hf,hb]@Wfch^T + bfch ; cell = [hf,hb]@Wfcc^T + bfcc
  gemm(stream, Bs, Hs, hFbf, Hs, wWfch, 2 * Hs, Hs, hBbf, Hs, wWfch + Hs, 2 * Hs, Hs,
       nullptr, 0, bfch, hD, Hs, 1, hDbf, Hs);
  gemm(stream, Bs, Hs, hFbf, Hs, wWfcc, 2 * Hs, Hs, hBbf, Hs, wWfcc + Hs, 2 * Hs, Hs,
       nullptr, 0, bfcc, cD, Hs, 1, nullptr, 0);

  // 9) attn_proj = enc_outputs @ Wattn^T + battn   (b*T rows, K=2H)
  gemm(stream, Bs * TSs, Hs, encbf, 2 * Hs, wWattn, 2 * Hs, 2 * Hs,
       nullptr, 0, nullptr, 0, 0, nullptr, 0, battn, attnp, Hs, 1, nullptr, 0);

  // 10) decoder loop
  for (int t = 0; t < TTs - 1; ++t) {
    // gates = XPd[t] + dout@Wih_d[:,H:2H]^T + h@Whh_d^T
    gemm(stream, Bs, Gs, doutbf, Hs, wWihD + Hs, 2 * Hs, Hs, hDbf, Hs, wWhhD, Hs, Hs,
         XPd + (size_t)t * Bs * Gs, Gs, nullptr, gates, Gs, 1, nullptr, 0);
    k_lstm_cell<<<cellBlocks, 256, 0, stream>>>(gates, nullptr, 0, hD, cD, hDbf,
                                                nullptr, nullptr, 0);
    // attention
    k_scores<<<dim3(TSs, Bs), 32, 0, stream>>>(attnp, hD, scores);
    k_softmax_mask<<<Bs, 64, 0, stream>>>(scores, src_len, wts);
    k_ctx<<<gblk(Bs * 2 * Hs), 256, 0, stream>>>(wts, encf, ctxbf);
    // dout = [ctx, h] @ Wcomb^T + bcomb   (f32 + bf16 feedback copy)
    gemm(stream, Bs, Hs, ctxbf, 2 * Hs, wWcomb, 3 * Hs, 2 * Hs,
         hDbf, Hs, wWcomb + 2 * Hs, 3 * Hs, Hs,
         nullptr, 0, bcomb, doutf, Hs, 1, doutbf, Hs);
    k_relu_bf16<<<gblk(Bs * Hs), 256, 0, stream>>>(doutf, doutrbf, Bs * Hs);
    // logits = relu(dout) @ Wout^T + bout -> d_out[b, v, t]  (strided epilogue)
    gemm(stream, Bs, Vs, doutrbf, Hs, wWout, Hs, Hs, nullptr, 0, nullptr, 0, 0,
         nullptr, 0, bout, out + t, (ll)Vs * (TTs - 1), (ll)(TTs - 1), nullptr, 0);
  }
}
